// Attention_13700945674459
// MI455X (gfx1250) — matrix operand order repaired and checked by reading, not compiled, not run on hardware
//
#include <hip/hip_runtime.h>

// ---------------------------------------------------------------------------
// Types
// ---------------------------------------------------------------------------
typedef unsigned short u16;
typedef __attribute__((ext_vector_type(16))) __bf16 bf16x16;
typedef __attribute__((ext_vector_type(8)))  float  f32x8;
typedef __attribute__((ext_vector_type(4)))  float  f32x4;
typedef __attribute__((ext_vector_type(4)))  unsigned short u16x4;

struct alignas(16) U128 { unsigned x, y, z, w; };
union FragU { bf16x16 v; U128 q[2]; };

static __device__ __forceinline__ u16 f2bf(float f) {
    unsigned u = __float_as_uint(f);
    unsigned r = u + 0x7fffu + ((u >> 16) & 1u);   // round-to-nearest-even
    return (u16)(r >> 16);
}
static __device__ __forceinline__ float bf2f(u16 b) {
    return __uint_as_float(((unsigned)b) << 16);
}

// 16-lane butterfly exchange done in VALU (v_permlane16_b32) instead of
// ds_bpermute: rows of 16 lanes match our half-wave row groups exactly.
static __device__ __forceinline__ float permxor16(float x, unsigned s0, unsigned s1) {
#if __has_builtin(__builtin_amdgcn_permlane16)
    unsigned u = __float_as_uint(x);
    unsigned r = __builtin_amdgcn_permlane16(u, u, s0, s1, false, false);
    return __uint_as_float(r);
#else
    (void)s0; (void)s1;
    return x;   // unreachable fallback replaced below
#endif
}

static __device__ __forceinline__ float rowmax16(float x) {
#if __has_builtin(__builtin_amdgcn_permlane16)
    x = fmaxf(x, permxor16(x, 0x67452301u, 0xEFCDAB89u)); // xor 1
    x = fmaxf(x, permxor16(x, 0x54761032u, 0xDCFE98BAu)); // xor 2
    x = fmaxf(x, permxor16(x, 0x32107654u, 0xBA98FEDCu)); // xor 4
    x = fmaxf(x, permxor16(x, 0xFEDCBA98u, 0x76543210u)); // xor 8
#else
    x = fmaxf(x, __shfl_xor(x, 1, 16));
    x = fmaxf(x, __shfl_xor(x, 2, 16));
    x = fmaxf(x, __shfl_xor(x, 4, 16));
    x = fmaxf(x, __shfl_xor(x, 8, 16));
#endif
    return x;
}
static __device__ __forceinline__ float rowsum16(float x) {
#if __has_builtin(__builtin_amdgcn_permlane16)
    x += permxor16(x, 0x67452301u, 0xEFCDAB89u);
    x += permxor16(x, 0x54761032u, 0xDCFE98BAu);
    x += permxor16(x, 0x32107654u, 0xBA98FEDCu);
    x += permxor16(x, 0xFEDCBA98u, 0x76543210u);
#else
    x += __shfl_xor(x, 1, 16);
    x += __shfl_xor(x, 2, 16);
    x += __shfl_xor(x, 4, 16);
    x += __shfl_xor(x, 8, 16);
#endif
    return x;
}

#define B_   4
#define NSEQ 2048
#define DMODEL 1024
#define HEADS 16
#define DH 64
#define SCALE 0.125f   // 64^-0.5

// ---------------------------------------------------------------------------
// f32 -> bf16 row-major convert (queries)
// ---------------------------------------------------------------------------
__global__ __launch_bounds__(256) void cvt_f32_bf16(const float* __restrict__ in,
                                                    u16* __restrict__ out) {
    size_t idx = ((size_t)blockIdx.x * 256 + threadIdx.x) * 4;
    f32x4 v = *(const f32x4*)(in + idx);
    u16x4 o;
    o[0] = f2bf(v[0]); o[1] = f2bf(v[1]); o[2] = f2bf(v[2]); o[3] = f2bf(v[3]);
    *(u16x4*)(out + idx) = o;
}

// ---------------------------------------------------------------------------
// f32 [K,N] -> bf16 [N,K]  (weights, so B-fragments load contiguously)
// ---------------------------------------------------------------------------
__global__ __launch_bounds__(256) void transpose_to_bf16(const float* __restrict__ in,
                                                         u16* __restrict__ out,
                                                         int K, int N) {
    __shared__ float tile[32][33];
    int tx = threadIdx.x, ty = threadIdx.y;
    int x0 = blockIdx.x * 32;   // N index
    int y0 = blockIdx.y * 32;   // K index
#pragma unroll
    for (int j = ty; j < 32; j += 8)
        tile[j][tx] = in[(size_t)(y0 + j) * N + (x0 + tx)];
    __syncthreads();
#pragma unroll
    for (int j = ty; j < 32; j += 8)
        out[(size_t)(x0 + j) * K + (y0 + tx)] = f2bf(tile[tx][j]);
}

// ---------------------------------------------------------------------------
// WMMA GEMM: C[M,N] = A[M,Kd] (bf16, row-major) x Bt[N,Kd] (bf16, N-major)
//   wave -> 16 rows x 64 cols (4 accumulators), block = 8 waves = 128 rows.
//   The 64x32 B tile is shared by all 8 waves -> staged into LDS once per
//   k-step via CDNA5 async DMA (global_load_async_to_lds_b128, ASYNCcnt),
//   ping-pong double buffered so the copy of tile k+1 overlaps WMMA of k.
//   MODE 0: f32 out + bias, row-major (final projection)
//   MODE 1: bf16 out packed [B,H,N,DH]           (Q projection)
//   MODE 2: n<1024 -> [B,H,N,DH] (K), n>=1024 -> [B,H,DH,N] (V transposed)
// ---------------------------------------------------------------------------
template <int MODE>
__global__ __launch_bounds__(256) void gemm_bf16(const u16* __restrict__ A,
                                                 const u16* __restrict__ Bt,
                                                 void* __restrict__ dst0,
                                                 void* __restrict__ dst1,
                                                 const float* __restrict__ bias,
                                                 int N, int Kd) {
    __shared__ __align__(16) u16 btile[2][64 * 32];   // 2 x 4KB ping-pong

    const int wave = threadIdx.x >> 5;
    const int lane = threadIdx.x & 31;
    const int lr = lane & 15;      // col within 16 / row within 16 (A)
    const int lh = lane >> 4;      // half-wave select
    const int m0 = blockIdx.x * 128 + wave * 16;
    const int n0 = blockIdx.y * 64;

    // async-copy assignment: thread -> 16B of the 64x32 bf16 B tile
    const int cCol = threadIdx.x >> 2;          // 0..63
    const int cK   = (threadIdx.x & 3) * 8;     // 0,8,16,24

    f32x8 acc[4];
    const f32x8 z8 = {0.f,0.f,0.f,0.f,0.f,0.f,0.f,0.f};
#pragma unroll
    for (int t = 0; t < 4; ++t) acc[t] = z8;

    const u16* aRow = A + (size_t)(m0 + lr) * Kd;
    const u16* bBase = Bt + (size_t)(n0 + cCol) * Kd + cK;

    // issue async copy of B tile for k-step k0 into buffer `buf`
    auto issue_b = [&](int k0, int buf) {
        unsigned lds = (unsigned)(size_t)&btile[buf][cCol * 32 + cK];
        unsigned long long g = (unsigned long long)(size_t)(bBase + k0);
        asm volatile("global_load_async_to_lds_b128 %0, %1, off"
                     :: "v"(lds), "v"(g) : "memory");
    };

    issue_b(0, 0);

    for (int k0 = 0; k0 < Kd; k0 += 32) {
        const int buf = (k0 >> 5) & 1;

        // wait for this k-step's tile (our own async), then block barrier:
        // (a) whole tile visible, (b) all waves finished reading buf^1.
        asm volatile("s_wait_asynccnt 0x0" ::: "memory");
        __syncthreads();
        if (k0 + 32 < Kd) issue_b(k0 + 32, buf ^ 1);
        if (k0 + 128 < Kd) __builtin_prefetch(aRow + k0 + 128, 0, 0);

        FragU af;
        af.q[0] = *(const U128*)(aRow + k0 + lh * 16);       // K = k0 + lh*16 + 0..7
        af.q[1] = *(const U128*)(aRow + k0 + lh * 16 + 8);   // K = k0 + lh*16 + 8..15
#pragma unroll
        for (int t = 0; t < 4; ++t) {
            const u16* bp = &btile[buf][(t * 16 + lr) * 32 + lh * 16];
            bf16x16 bf = *(const bf16x16*)bp;                // K = k0 + lh*16 + 0..15
            acc[t] = __builtin_amdgcn_wmma_f32_16x16x32_bf16(
                         false, af.v, false, bf, (short)0, acc[t], false, false);
        }
    }

#pragma unroll
    for (int t = 0; t < 4; ++t) {
#pragma unroll
        for (int r = 0; r < 8; ++r) {
            int m = m0 + lh * 8 + r;          // C layout: lanes>=16 hold rows 8..15
            int n = n0 + t * 16 + lr;
            float v = acc[t][r];
            if (MODE == 0) {
                ((float*)dst0)[(size_t)m * N + n] = v + bias[n];
            } else if (MODE == 1) {
                int b = m >> 11, i = m & 2047;
                int h = n >> 6,  dd = n & 63;
                ((u16*)dst0)[(((size_t)b * HEADS + h) * NSEQ + i) * DH + dd] = f2bf(v);
            } else {
                int b = m >> 11, i = m & 2047;
                if (n < DMODEL) {
                    int h = n >> 6, dd = n & 63;
                    ((u16*)dst0)[(((size_t)b * HEADS + h) * NSEQ + i) * DH + dd] = f2bf(v);
                } else {
                    int n2 = n - DMODEL;
                    int h = n2 >> 6, dd = n2 & 63;
                    ((u16*)dst1)[(((size_t)b * HEADS + h) * DH + dd) * NSEQ + i] = f2bf(v);
                }
            }
        }
    }
}

// ---------------------------------------------------------------------------
// RoPE in place on bf16 [B*H, NSEQ, DH]; rotates dims 0..31 in pairs (2t,2t+1)
// ---------------------------------------------------------------------------
__global__ __launch_bounds__(256) void rope_kernel(u16* __restrict__ buf) {
    size_t tid = (size_t)blockIdx.x * 256 + threadIdx.x;
    int t = (int)(tid & 15);
    size_t row = tid >> 4;                         // 0 .. B*H*NSEQ-1
    int pos = (int)(row & (NSEQ - 1));
    float inv = __powf(10000.f, -(float)t / 16.f); // 10000^{-2t/32}
    float ang = (float)pos * inv;
    float s, c;
    __sincosf(ang, &s, &c);
    u16* p = buf + row * DH + 2 * t;
    float x1 = bf2f(p[0]), x2 = bf2f(p[1]);
    p[0] = f2bf(x1 * c - x2 * s);
    p[1] = f2bf(x2 * c + x1 * s);
}

// ---------------------------------------------------------------------------
// Flash attention: one wave per 16-query tile; 4 waves/block.
//   Q,K : bf16 [B,H,NSEQ,DH]     V : bf16 [B,H,DH,NSEQ] (transposed)
//   O   : bf16 [B*NSEQ, H*DH]
// Per 32-key step: 4 WMMA (S = Q K^T), online softmax with v_permlane16
// row reductions, P routed through LDS (C-layout f32 -> A-layout bf16),
// 4 WMMA (O += P V).
// ---------------------------------------------------------------------------
__global__ __launch_bounds__(128) void attn_kernel(const u16* __restrict__ Q,
                                                   const u16* __restrict__ K,
                                                   const u16* __restrict__ Vt,
                                                   u16* __restrict__ O) {
    __shared__ __align__(16) u16 pTile[4][16 * 32];

    const int wave = threadIdx.x >> 5;
    const int lane = threadIdx.x & 31;
    const int lr = lane & 15;
    const int lh = lane >> 4;

    const int tile = blockIdx.x * 4 + wave;
    const int it = tile & 127;          // query tile within sequence (NSEQ/16)
    const int bh = tile >> 7;           // 0 .. B*HEADS-1
    const int h = bh & 15, b = bh >> 4;
    const int i0 = it * 16;

    const u16* Qb = Q + ((size_t)bh * NSEQ + i0) * DH;
    const u16* Kb = K + (size_t)bh * NSEQ * DH;
    const u16* Vb = Vt + (size_t)bh * DH * NSEQ;
    u16* pt = &pTile[wave][0];

    // Q fragments for K-dim steps 0 and 32 (A-layout)
    FragU qf[2];
#pragma unroll
    for (int kk = 0; kk < 2; ++kk) {
        const u16* base = Qb + (size_t)lr * DH + kk * 32 + lh * 16;
        qf[kk].q[0] = *(const U128*)(base);
        qf[kk].q[1] = *(const U128*)(base + 8);
    }

    const f32x8 z8 = {0.f,0.f,0.f,0.f,0.f,0.f,0.f,0.f};
    f32x8 o[4];
#pragma unroll
    for (int t = 0; t < 4; ++t) o[t] = z8;
    float mrow[8], lrow[8];
#pragma unroll
    for (int r = 0; r < 8; ++r) { mrow[r] = -3.0e38f; lrow[r] = 0.f; }

    for (int j0 = 0; j0 < NSEQ; j0 += 32) {
        // ---- S = Q K^T for key cols [j0, j0+32) --------------------------
        f32x8 s0 = z8, s1 = z8;
#pragma unroll
        for (int kk = 0; kk < 2; ++kk) {
            const u16* kp0 = Kb + (size_t)(j0 + lr) * DH + kk * 32 + lh * 16;
            bf16x16 kb0 = *(const bf16x16*)kp0;
            bf16x16 kb1 = *(const bf16x16*)(kp0 + 16 * DH);
            s0 = __builtin_amdgcn_wmma_f32_16x16x32_bf16(false, qf[kk].v, false, kb0,
                                                         (short)0, s0, false, false);
            s1 = __builtin_amdgcn_wmma_f32_16x16x32_bf16(false, qf[kk].v, false, kb1,
                                                         (short)0, s1, false, false);
        }

        // ---- online softmax (rows live on 16 lanes of a half-wave) -------
        float p0[8], p1[8];
#pragma unroll
        for (int r = 0; r < 8; ++r) {
            float a0 = s0[r] * SCALE, a1 = s1[r] * SCALE;
            float mx = rowmax16(fmaxf(a0, a1));
            float mnew = fmaxf(mrow[r], mx);
            float alpha = __expf(mrow[r] - mnew);
            mrow[r] = mnew;
            p0[r] = __expf(a0 - mnew);
            p1[r] = __expf(a1 - mnew);
            float sm = rowsum16(p0[r] + p1[r]);
            lrow[r] = lrow[r] * alpha + sm;
            o[0][r] *= alpha; o[1][r] *= alpha; o[2][r] *= alpha; o[3][r] *= alpha;
        }

        // ---- re-layout P: C-format f32 -> A-format bf16 via LDS ----------
#pragma unroll
        for (int r = 0; r < 8; ++r) {
            int rowp = lh * 8 + r;
            pt[rowp * 32 + lr]      = f2bf(p0[r]);
            pt[rowp * 32 + 16 + lr] = f2bf(p1[r]);
        }
        asm volatile("s_wait_dscnt 0" ::: "memory");
        FragU pa;
        pa.q[0] = *(const U128*)(pt + lr * 32 + lh * 16);
        pa.q[1] = *(const U128*)(pt + lr * 32 + lh * 16 + 8);

        // ---- O += P V  (V transposed -> contiguous B-fragments) ----------
#pragma unroll
        for (int t = 0; t < 4; ++t) {
            const u16* vp = Vb + (size_t)(t * 16 + lr) * NSEQ + j0 + lh * 16;
            bf16x16 bv = *(const bf16x16*)vp;
            o[t] = __builtin_amdgcn_wmma_f32_16x16x32_bf16(false, pa.v, false, bv,
                                                           (short)0, o[t], false, false);
        }
    }

    // ---- normalize and store to [B*NSEQ, H*DH] ---------------------------
#pragma unroll
    for (int r = 0; r < 8; ++r) {
        float inv_l = 1.0f / lrow[r];
        size_t rowOut = (size_t)b * NSEQ + i0 + lh * 8 + r;
#pragma unroll
        for (int t = 0; t < 4; ++t)
            O[rowOut * (HEADS * DH) + h * DH + t * 16 + lr] = f2bf(o[t][r] * inv_l);
    }
}

// ---------------------------------------------------------------------------
// Host launcher
// ---------------------------------------------------------------------------
extern "C" void kernel_launch(void* const* d_in, const int* in_sizes, int n_in,
                              void* d_out, int out_size, void* d_ws, size_t ws_size,
                              hipStream_t stream) {
    const float* q_in  = (const float*)d_in[0];   // [4,2048,1024]
    const float* Wq    = (const float*)d_in[1];   // [1024,1024]
    const float* Wkv   = (const float*)d_in[2];   // [1024,2048]
    const float* Wout  = (const float*)d_in[3];   // [1024,1024]
    const float* bout  = (const float*)d_in[4];   // [1024]
    float* out = (float*)d_out;

    char* ws = (char*)d_ws;
    size_t off = 0;
    auto alloc = [&](size_t bytes) -> void* {
        off = (off + 255) & ~(size_t)255;
        void* p = ws + off;
        off += bytes;
        return p;
    };
    const size_t actElems = (size_t)B_ * NSEQ * DMODEL;   // 8192*1024
    u16* Xbf   = (u16*)alloc(actElems * 2);
    u16* WqT   = (u16*)alloc((size_t)DMODEL * DMODEL * 2);
    u16* WkvT  = (u16*)alloc((size_t)2 * DMODEL * DMODEL * 2);
    u16* WoutT = (u16*)alloc((size_t)DMODEL * DMODEL * 2);
    u16* Qb    = (u16*)alloc(actElems * 2);               // [B,H,N,DH]
    u16* Kb    = (u16*)alloc(actElems * 2);               // [B,H,N,DH]
    u16* Vt    = (u16*)alloc(actElems * 2);               // [B,H,DH,N]
    u16* Ao    = (u16*)alloc(actElems * 2);               // [B*N, H*DH]
    (void)ws_size; (void)in_sizes; (void)n_in; (void)out_size;

    // 1. activations -> bf16
    cvt_f32_bf16<<<(unsigned)(actElems / (256 * 4)), 256, 0, stream>>>(q_in, Xbf);

    // 2. weights -> bf16, transposed to [N,K]
    transpose_to_bf16<<<dim3(DMODEL / 32, DMODEL / 32), dim3(32, 8), 0, stream>>>(Wq, WqT, DMODEL, DMODEL);
    transpose_to_bf16<<<dim3(2 * DMODEL / 32, DMODEL / 32), dim3(32, 8), 0, stream>>>(Wkv, WkvT, DMODEL, 2 * DMODEL);
    transpose_to_bf16<<<dim3(DMODEL / 32, DMODEL / 32), dim3(32, 8), 0, stream>>>(Wout, WoutT, DMODEL, DMODEL);

    const int Mrows = B_ * NSEQ;  // 8192
    // 3. Q projection -> [B,H,N,DH]
    gemm_bf16<1><<<dim3(Mrows / 128, DMODEL / 64), 256, 0, stream>>>(Xbf, WqT, Qb, nullptr, nullptr, DMODEL, DMODEL);
    // 4. KV projection -> K [B,H,N,DH], V [B,H,DH,N]
    gemm_bf16<2><<<dim3(Mrows / 128, 2 * DMODEL / 64), 256, 0, stream>>>(Xbf, WkvT, Kb, Vt, nullptr, 2 * DMODEL, DMODEL);

    // 5. RoPE on Q and K
    const unsigned ropeBlocks = (unsigned)((size_t)B_ * HEADS * NSEQ * 16 / 256);
    rope_kernel<<<ropeBlocks, 256, 0, stream>>>(Qb);
    rope_kernel<<<ropeBlocks, 256, 0, stream>>>(Kb);

    // 6. attention: B*H*(N/16) tiles, 4 waves/block
    attn_kernel<<<(unsigned)(B_ * HEADS * (NSEQ / 16) / 4), 128, 0, stream>>>(Qb, Kb, Vt, Ao);

    // 7. output projection (f32 + bias) -> d_out
    gemm_bf16<0><<<dim3(Mrows / 128, DMODEL / 64), 256, 0, stream>>>(Ao, WoutT, out, nullptr, bout, DMODEL, DMODEL);
}
